// AutoEncoder_13280038879993
// MI455X (gfx1250) — matrix operand order repaired and checked by reading, not compiled, not run on hardware
//
#include <hip/hip_runtime.h>

// LSTM autoencoder for MI455X (gfx1250, wave32, WMMA bf16 16x16x32).
// Grid: 16 WGs x 256 threads (8 waves). Each WG owns a 16-row batch tile and
// runs the full sequential recurrence with weights pre-swizzled to bf16 WMMA
// B-fragment layout in LDS. f32 accumulation throughout.
// v2: x_t staged through registers one step ahead (global-load latency hidden
// behind a full LSTM step); loop-invariant biases hoisted out of the t-loop.

#define RS 100            // panel row stride in dwords (bf16 pairs), padded vs 96
#define RS2 (RS * 2)      // panel row stride in ushorts
#define TSTEPS 512
#define BT 16             // batch tile per workgroup

typedef __attribute__((ext_vector_type(16))) __bf16        v16bf;
typedef __attribute__((ext_vector_type(8)))  float         v8f;
typedef __attribute__((ext_vector_type(4)))  unsigned int  u32x4;
typedef __attribute__((ext_vector_type(4)))  float         f32x4;

union FragCast { u32x4 u[2]; v16bf b; };

__device__ __forceinline__ unsigned short f2bf(float f) {
  unsigned int u = __float_as_uint(f);
  unsigned int r = ((u >> 16) & 1u) + 0x7FFFu;   // round-to-nearest-even
  return (unsigned short)((u + r) >> 16);
}

__device__ __forceinline__ float sigmoidf_(float x) {
  return 1.0f / (1.0f + __expf(-x));
}

// A-fragment (16x32 bf16, ISA layout): lanes 0-15 = M, K pairs 0..7 + 16..23;
// lanes 16-31 = M, K pairs 8..15 + 24..31. Panel holds bf16 contiguous in K.
__device__ __forceinline__ v16bf load_a_frag(const unsigned int* panel, int colDwBase,
                                             int kc, int lane) {
  int m    = lane & 15;
  int half = lane >> 4;
  const unsigned int* p = panel + m * RS + colDwBase + kc * 16 + half * 4;
  FragCast fc;
  fc.u[0] = *(const u32x4*)(p);
  fc.u[1] = *(const u32x4*)(p + 8);
  return fc.b;
}

// B-fragment (32x16 bf16): pre-swizzled, 8 contiguous dwords per lane per frag.
__device__ __forceinline__ v16bf load_b_frag(const unsigned int* frags, int NT,
                                             int kc, int nt, int lane) {
  const unsigned int* p = frags + (((kc * NT) + nt) * 32 + lane) * 8;
  FragCast fc;
  fc.u[0] = *(const u32x4*)(p);
  fc.u[1] = *(const u32x4*)(p + 4);
  return fc.b;
}

__device__ __forceinline__ v8f wmma_bf16(v16bf a, v16bf b, v8f c) {
  return __builtin_amdgcn_wmma_f32_16x16x32_bf16(false, a, false, b, (short)0, c,
                                                 false, false);
}

// One LSTM cell step for this wave's 16-wide h-tile. Gate tiles i/f/g/o share
// lane/slot positions across 4 accumulators -> pure VALU gate math.
// bias4 = {bi, bf, bg, bo} hoisted to registers by the caller (loop-invariant).
template <int KC, int HT>
__device__ __forceinline__ v8f lstm_tile(int lane, int htile,
                                         const unsigned int* panel,
                                         const unsigned int* wfrags,
                                         f32x4 bias4, v8f& c) {
  const int NT = 4 * HT;
  v8f zi, zf, zg, zo;
#pragma unroll
  for (int r = 0; r < 8; ++r) {
    zi[r] = bias4.x; zf[r] = bias4.y; zg[r] = bias4.z; zo[r] = bias4.w;
  }
#pragma unroll
  for (int kc = 0; kc < KC; ++kc) {
    v16bf a = load_a_frag(panel, 0, kc, lane);
    zi = wmma_bf16(a, load_b_frag(wfrags, NT, kc, htile, lane), zi);
    zf = wmma_bf16(a, load_b_frag(wfrags, NT, kc, HT + htile, lane), zf);
    zg = wmma_bf16(a, load_b_frag(wfrags, NT, kc, 2 * HT + htile, lane), zg);
    zo = wmma_bf16(a, load_b_frag(wfrags, NT, kc, 3 * HT + htile, lane), zo);
  }
  v8f h;
#pragma unroll
  for (int r = 0; r < 8; ++r) {
    float iv = sigmoidf_(zi[r]);
    float fv = sigmoidf_(zf[r]);
    float gv = fmaxf(zg[r], 0.0f);     // activation='relu'
    float ov = sigmoidf_(zo[r]);
    float cv = fv * c[r] + iv * gv;
    c[r] = cv;
    h[r] = ov * fmaxf(cv, 0.0f);       // relu on cell output
  }
  return h;
}

// Gate biases for this wave's h-tile, loaded once from LDS into registers.
__device__ __forceinline__ f32x4 load_bias4(const float* bias, int htile, int HT,
                                            int lane) {
  int n = lane & 15;
  f32x4 b;
  b.x = bias[htile * 16 + n];
  b.y = bias[(HT + htile) * 16 + n];
  b.z = bias[(2 * HT + htile) * 16 + n];
  b.w = bias[(3 * HT + htile) * 16 + n];
  return b;
}

// Store h tile as bf16 into one or two panels (dA/dB pre-offset by col base).
__device__ __forceinline__ void store_h(v8f h, int lane, int htile,
                                        unsigned short* dA, unsigned short* dB) {
  int n  = htile * 16 + (lane & 15);
  int mb = (lane >> 4) * 8;
#pragma unroll
  for (int r = 0; r < 8; ++r) {
    unsigned short v = f2bf(h[r]);
    int m = mb + r;
    dA[m * RS2 + n] = v;
    if (dB) dB[m * RS2 + n] = v;
  }
}

// Convert [Wk;Wr] (f32, row-major KxN) to bf16 B-fragments in LDS.
__device__ void swizzle_weights(const float* Wk, int Din, const float* Wr,
                                int Ktot, int N, unsigned int* dst,
                                int tid, int nthreads) {
  int NT = N / 16;
  int total = (Ktot / 32) * NT * 256;          // dwords
  for (int idx = tid; idx < total; idx += nthreads) {
    int r    = idx & 7;
    int lane = (idx >> 3) & 31;
    int frag = idx >> 8;
    int kc   = frag / NT;
    int nt   = frag - kc * NT;
    int n  = nt * 16 + (lane & 15);
    int k0 = kc * 32 + (lane >> 4) * 8 + 2 * r + (r >= 4 ? 8 : 0);
    int k1 = k0 + 1;
    float w0 = (k0 < Din) ? Wk[(size_t)k0 * N + n] : Wr[(size_t)(k0 - Din) * N + n];
    float w1 = (k1 < Din) ? Wk[(size_t)k1 * N + n] : Wr[(size_t)(k1 - Din) * N + n];
    dst[idx] = (unsigned int)f2bf(w0) | ((unsigned int)f2bf(w1) << 16);
  }
}

// ---------------- Encoder: L1 (F=64 -> H1=128) + L2 (128 -> H2=64) ----------
extern "C" __global__ void __launch_bounds__(256, 1)
lstm_encoder_kernel(const float* __restrict__ x,
                    const float* __restrict__ Wk1, const float* __restrict__ Wr1,
                    const float* __restrict__ b1,
                    const float* __restrict__ Wk2, const float* __restrict__ Wr2,
                    const float* __restrict__ b2,
                    unsigned short* __restrict__ zbuf) {
  extern __shared__ __align__(16) unsigned int smem[];
  unsigned int* W1f = smem;                      // KC=6, NT=32 -> 49152 dw
  unsigned int* W2f = W1f + 49152;               // KC=6, NT=16 -> 24576 dw
  float* b1s        = (float*)(W2f + 24576);     // 512
  float* b2s        = b1s + 512;                 // 256
  unsigned int* p1  = (unsigned int*)(b2s + 256); // panel1: [x_t | h1], 16*RS dw
  unsigned int* p2  = p1 + BT * RS;               // panel2: [h1 | h2]

  const int tid = threadIdx.x;
  swizzle_weights(Wk1, 64, Wr1, 192, 512, W1f, tid, 256);
  swizzle_weights(Wk2, 128, Wr2, 192, 256, W2f, tid, 256);
  for (int i = tid; i < 512; i += 256) b1s[i] = b1[i];
  for (int i = tid; i < 256; i += 256) b2s[i] = b2[i];
  for (int i = tid; i < BT * RS; i += 256) { p1[i] = 0u; p2[i] = 0u; }
  __syncthreads();

  const int w    = tid >> 5;
  const int lane = tid & 31;
  const int b0   = blockIdx.x * BT;

  const f32x4 bias1 = load_bias4(b1s, w, 8, lane);
  const f32x4 bias2 = load_bias4(b2s, w & 3, 4, lane);

  v8f c1, c2, h2;
#pragma unroll
  for (int r = 0; r < 8; ++r) { c1[r] = 0.f; c2[r] = 0.f; h2[r] = 0.f; }

  // x staging: each thread owns 4 consecutive f32 of the 16x64 x_t tile.
  const int xm   = (tid * 4) >> 6;
  const int xcol = (tid * 4) & 63;
  const float* xbase = x + ((size_t)(b0 + xm) * TSTEPS) * 64 + xcol;
  f32x4 xreg = *(const f32x4*)(xbase);           // prefetch x_0

  for (int t = 0; t < TSTEPS; ++t) {
    {  // commit x_t (held in registers) to panel1 cols 0..63 as bf16
      p1[xm * RS + (xcol >> 1)]     = (unsigned int)f2bf(xreg.x) | ((unsigned int)f2bf(xreg.y) << 16);
      p1[xm * RS + (xcol >> 1) + 1] = (unsigned int)f2bf(xreg.z) | ((unsigned int)f2bf(xreg.w) << 16);
      int tn = (t + 1 < TSTEPS) ? t + 1 : t;     // prefetch x_{t+1} (clamped)
      xreg = *(const f32x4*)(xbase + (size_t)tn * 64);
    }
    __syncthreads();

    v8f h1 = lstm_tile<6, 8>(lane, w, p1, W1f, bias1, c1);   // all 8 waves
    __syncthreads();
    store_h(h1, lane, w, (unsigned short*)p1 + 64, (unsigned short*)p2);
    __syncthreads();

    if (w < 4) h2 = lstm_tile<6, 4>(lane, w, p2, W2f, bias2, c2);
    __syncthreads();
    if (w < 4) store_h(h2, lane, w, (unsigned short*)p2 + 128, nullptr);
    __syncthreads();
  }

  {  // export z = final h2 (bf16) for the decoder
    int e = tid * 4;
    int m = e >> 6;
    int col = e & 63;
#pragma unroll
    for (int j = 0; j < 4; ++j)
      zbuf[(size_t)(b0 + m) * 64 + col + j] =
          ((unsigned short*)p2)[m * RS2 + 128 + col + j];
  }
}

// ------------- Decoder: L3 (64 -> 64) + L4 (64 -> 128) + Dense (128 -> 64) --
extern "C" __global__ void __launch_bounds__(256, 1)
lstm_decoder_kernel(const unsigned short* __restrict__ zbuf,
                    const float* __restrict__ Wd1k, const float* __restrict__ Wd1r,
                    const float* __restrict__ bd1,
                    const float* __restrict__ Wd2k, const float* __restrict__ Wd2r,
                    const float* __restrict__ bd2,
                    const float* __restrict__ Wout, const float* __restrict__ bout,
                    float* __restrict__ out) {
  extern __shared__ __align__(16) unsigned int smem[];
  unsigned int* W3f = smem;                       // KC=4, NT=16 -> 16384 dw
  unsigned int* W4f = W3f + 16384;                // KC=6, NT=32 -> 49152 dw
  unsigned int* Wdf = W4f + 49152;                // KC=4, NT=4  -> 4096 dw
  float* b3s        = (float*)(Wdf + 4096);       // 256
  float* b4s        = b3s + 256;                  // 512
  float* bos        = b4s + 512;                  // 64
  unsigned int* p3  = (unsigned int*)(bos + 64);  // panel3: [z | h3]
  unsigned int* p4  = p3 + BT * RS;               // panel4: [h3 | h4]

  const int tid = threadIdx.x;
  const int b0  = blockIdx.x * BT;
  swizzle_weights(Wd1k, 64, Wd1r, 128, 256, W3f, tid, 256);
  swizzle_weights(Wd2k, 64, Wd2r, 192, 512, W4f, tid, 256);
  swizzle_weights(Wout, 128, nullptr, 128, 64, Wdf, tid, 256);
  for (int i = tid; i < 256; i += 256) b3s[i] = bd1[i];
  for (int i = tid; i < 512; i += 256) b4s[i] = bd2[i];
  if (tid < 64) bos[tid] = bout[tid];
  for (int i = tid; i < BT * RS; i += 256) { p3[i] = 0u; p4[i] = 0u; }
  __syncthreads();

  {  // load constant z (RepeatVector input) into panel3 cols 0..63
    int e = tid * 4;
    int m = e >> 6;
    int col = e & 63;
#pragma unroll
    for (int j = 0; j < 4; ++j)
      ((unsigned short*)p3)[m * RS2 + col + j] = zbuf[(size_t)(b0 + m) * 64 + col + j];
  }
  __syncthreads();

  const int w    = tid >> 5;
  const int lane = tid & 31;

  const f32x4 bias3 = load_bias4(b3s, w & 3, 4, lane);
  const f32x4 bias4v = load_bias4(b4s, w, 8, lane);
  const float biasd = bos[(w & 3) * 16 + (lane & 15)];

  v8f c3, c4, h3;
#pragma unroll
  for (int r = 0; r < 8; ++r) { c3[r] = 0.f; c4[r] = 0.f; h3[r] = 0.f; }

  for (int t = 0; t < TSTEPS; ++t) {
    if (w < 4) h3 = lstm_tile<4, 4>(lane, w, p3, W3f, bias3, c3);
    __syncthreads();
    if (w < 4) store_h(h3, lane, w, (unsigned short*)p3 + 64, (unsigned short*)p4);
    __syncthreads();

    v8f h4 = lstm_tile<6, 8>(lane, w, p4, W4f, bias4v, c4);  // all 8 waves
    __syncthreads();
    store_h(h4, lane, w, (unsigned short*)p4 + 64, nullptr);
    __syncthreads();

    if (w < 4) {  // TimeDistributed Dense on h4 (panel4 cols 64..191)
      int n = lane & 15;
      v8f acc;
#pragma unroll
      for (int r = 0; r < 8; ++r) acc[r] = biasd;
#pragma unroll
      for (int kc = 0; kc < 4; ++kc) {
        v16bf a = load_a_frag(p4, 32, kc, lane);
        acc = wmma_bf16(a, load_b_frag(Wdf, 4, kc, w, lane), acc);
      }
      int mb = (lane >> 4) * 8;
#pragma unroll
      for (int r = 0; r < 8; ++r) {
        int m = mb + r;
        out[(((size_t)(b0 + m)) * TSTEPS + t) * 64 + w * 16 + n] = acc[r];
      }
    }
  }
}

extern "C" void kernel_launch(void* const* d_in, const int* in_sizes, int n_in,
                              void* d_out, int out_size, void* d_ws, size_t ws_size,
                              hipStream_t stream) {
  const float* x    = (const float*)d_in[0];
  const float* Wk1  = (const float*)d_in[1];
  const float* Wr1  = (const float*)d_in[2];
  const float* b1   = (const float*)d_in[3];
  const float* Wk2  = (const float*)d_in[4];
  const float* Wr2  = (const float*)d_in[5];
  const float* b2   = (const float*)d_in[6];
  const float* Wd1k = (const float*)d_in[7];
  const float* Wd1r = (const float*)d_in[8];
  const float* bd1  = (const float*)d_in[9];
  const float* Wd2k = (const float*)d_in[10];
  const float* Wd2r = (const float*)d_in[11];
  const float* bd2  = (const float*)d_in[12];
  const float* Wout = (const float*)d_in[13];
  const float* bout = (const float*)d_in[14];
  float* out = (float*)d_out;
  unsigned short* zbuf = (unsigned short*)d_ws;   // 256*64 bf16 = 32 KB

  constexpr size_t encLds =
      (size_t)(49152 + 24576) * 4 + (512 + 256) * 4 + (size_t)2 * BT * RS * 4;  // ~304 KB
  constexpr size_t decLds =
      (size_t)(16384 + 49152 + 4096) * 4 + (256 + 512 + 64) * 4 + (size_t)2 * BT * RS * 4;  // ~288 KB

  (void)hipFuncSetAttribute((const void*)lstm_encoder_kernel,
                            hipFuncAttributeMaxDynamicSharedMemorySize, (int)encLds);
  (void)hipFuncSetAttribute((const void*)lstm_decoder_kernel,
                            hipFuncAttributeMaxDynamicSharedMemorySize, (int)decLds);

  lstm_encoder_kernel<<<dim3(16), dim3(256), encLds, stream>>>(x, Wk1, Wr1, b1,
                                                               Wk2, Wr2, b2, zbuf);
  lstm_decoder_kernel<<<dim3(16), dim3(256), decLds, stream>>>(zbuf, Wd1k, Wd1r, bd1,
                                                               Wd2k, Wd2r, bd2,
                                                               Wout, bout, out);
}
